// GAT_Pure_91268055040316
// MI455X (gfx1250) — matrix operand order repaired and checked by reading, compiled, not run on hardware
//
#include <hip/hip_runtime.h>

#define NN 50000
#define NE 300000
#define RT 3125           // 50000 / 16 row tiles (exact)

typedef __attribute__((ext_vector_type(16))) __bf16 v16bf;
typedef __attribute__((ext_vector_type(8)))  float  v8f;
typedef __attribute__((ext_vector_type(4)))  unsigned int u32x4;
typedef __attribute__((ext_vector_type(8)))  int i32x8;
typedef __attribute__((ext_vector_type(4)))  int i32x4;

union BfVec { uint4 q[2]; v16bf v; };

__device__ __forceinline__ float bf2f(unsigned h) {            // h: low 16 bits valid
    return __uint_as_float(h << 16);
}
__device__ __forceinline__ unsigned short f2bf(float f) {      // round-to-nearest-even
    unsigned u = __float_as_uint(f);
    u += 0x7FFFu + ((u >> 16) & 1u);
    return (unsigned short)(u >> 16);
}
__device__ __forceinline__ unsigned f2ord(float f) {           // sortable-uint encode
    unsigned u = __float_as_uint(f);
    return (u & 0x80000000u) ? ~u : (u | 0x80000000u);
}
__device__ __forceinline__ float ord2f(unsigned u) {
    unsigned v = (u & 0x80000000u) ? (u & 0x7FFFFFFFu) : ~u;
    return __uint_as_float(v);
}

// ---------------------------------------------------------------------------
// Generic fp32->bf16 convert
// ---------------------------------------------------------------------------
__global__ void cvt_f32_bf16(const float* __restrict__ in,
                             unsigned short* __restrict__ out, int n) {
    int i = blockIdx.x * blockDim.x + threadIdx.x;
    if (i < n) out[i] = f2bf(in[i]);
}

// Transpose-convert weight W[K,M] (row-major fp32) -> Wt[Mpad,K] (row-major bf16),
// zero-padding columns M..Mpad-1.
__global__ void wcvt_t(const float* __restrict__ W, unsigned short* __restrict__ Wt,
                       int K, int M, int Mpad) {
    int i = blockIdx.x * blockDim.x + threadIdx.x;
    if (i >= Mpad * K) return;
    int n = i / K, k = i - n * K;
    float v = (n < M) ? W[(size_t)k * M + n] : 0.0f;
    Wt[i] = f2bf(v);
}

// ---------------------------------------------------------------------------
// Stage Wt[Mpad*K] bf16 into LDS.  Preferred path: Tensor Data Mover (1-D
// descriptor, data_size=2B), issued by wave 0 + s_wait_tensorcnt + barrier.
// This toolchain exposes the 6-arg builtin:
//   (u32x4 g0, i32x8 g1, i32x4 g2, i32x4 g3, i32x8 extra, i32 cpol)
// Fallback: cooperative b128 copy.
// ---------------------------------------------------------------------------
__device__ __forceinline__ void stage_weights_lds(unsigned short* lWt,
                                                  const unsigned short* Wt,
                                                  int nElems, int wave) {
#if __has_builtin(__builtin_amdgcn_tensor_load_to_lds) && \
    __has_builtin(__builtin_amdgcn_s_wait_tensorcnt)
    if (wave == 0) {
        unsigned long long ga = (unsigned long long)(uintptr_t)Wt;
        // D# group 0: count=1 | lds_addr=0 | global_addr | type=2 ("image")
        u32x4 g0;
        g0.x = 1u;                                        // count=1 (valid)
        g0.y = 0u;                                        // lds_addr = 0 (dyn-LDS base)
        g0.z = (unsigned)(ga & 0xFFFFFFFFu);              // global_addr[31:0]
        g0.w = (unsigned)((ga >> 32) & 0x01FFFFFFu) | (2u << 30);  // [56:32] | type=2
        // D# group 1: mask=0, data_size=1(2B), 1-D tensor of nElems, tile=nElems
        i32x8 g1;
        g1[0] = (1 << 16);                                // data_size code 1 -> 2 bytes
        g1[1] = (int)((nElems & 0xFFFF) << 16);           // tensor_dim0[15:0]
        g1[2] = (int)(((unsigned)nElems >> 16) & 0xFFFFu) // tensor_dim0[31:16]
              | (1 << 16);                                // tensor_dim1 = 1
        g1[3] = (int)((nElems & 0xFFFF) << 16);           // tile_dim0 = nElems
        g1[4] = 0;                                        // tile_dim1/2 unused
        g1[5] = nElems;                                   // tensor_dim0_stride[31:0]
        g1[6] = 0;
        g1[7] = 0;
        i32x4 z4 = {0, 0, 0, 0};
        i32x8 z8 = {0, 0, 0, 0, 0, 0, 0, 0};
        __builtin_amdgcn_tensor_load_to_lds(g0, g1, z4, z4, z8, 0);
        __builtin_amdgcn_s_wait_tensorcnt(0);
    }
    __syncthreads();
#else
    int tid = threadIdx.x;
    uint4* d = (uint4*)lWt;
    const uint4* s = (const uint4*)Wt;
    for (int i = tid; i < (nElems >> 3); i += blockDim.x) d[i] = s[i];
    __syncthreads();
#endif
    (void)wave;
}

// ---------------------------------------------------------------------------
// bf16 WMMA GEMM:  out[N, Mreal] = act(A[N,K] @ Wt^T + bias)
//   A   : bf16 row-major [NN, K]
//   Wt  : bf16 row-major [Mtiles*16, K]   (transposed weights, possibly padded)
//   Wt is staged in LDS (TDM); each wave computes 16 rows x CT*16 cols so one
//   A fragment feeds CT WMMAs per k-step.  8 waves / 256-thread block.
// ---------------------------------------------------------------------------
template <int CT>
__global__ void gemm_bf16_wmma(const unsigned short* __restrict__ A,
                               const unsigned short* __restrict__ Wt,
                               const float* __restrict__ bias,
                               unsigned short* __restrict__ outb,
                               float* __restrict__ out32,
                               int K, int Mtiles, int Mreal, int relu) {
    extern __shared__ __align__(16) unsigned short lWt[];
    int wave = threadIdx.x >> 5;
    int lane = threadIdx.x & 31;

    stage_weights_lds(lWt, Wt, Mtiles * 16 * K, wave);

    int colGroups = Mtiles / CT;
    int tile = blockIdx.x * (blockDim.x >> 5) + wave;
    int total = RT * colGroups;
    if (tile >= total) return;                       // wave-uniform
    int rowTile = tile / colGroups;
    int cg = tile - rowTile * colGroups;
    int hi = lane >> 4;                              // 0: lanes 0-15, 1: lanes 16-31
    int aRow = rowTile * 16 + (lane & 15);
    int nCol0 = cg * CT * 16 + (lane & 15);

    const unsigned short* aBase = A + (size_t)aRow * K;
    __builtin_prefetch(aBase, 0, 1);

    v8f zero = {};
    v8f acc[CT];
#pragma unroll
    for (int t = 0; t < CT; ++t) acc[t] = zero;

    for (int kk = 0; kk < K; kk += 32) {
        // A 16x32 bf16 layout: lanes 0-15 hold K kk..kk+7 & kk+16..kk+23;
        //                      lanes 16-31 hold K kk+8..kk+15 & kk+24..kk+31.
        int kA = kk + hi * 8;
        BfVec av;
        av.q[0] = *(const uint4*)(aBase + kA);
        av.q[1] = *(const uint4*)(aBase + kA + 16);
        // B 32x16 bf16 layout: same K order as A (lanes 0-15 kk..kk+7 & kk+16..kk+23; lanes 16-31 kk+8..kk+15 & kk+24..kk+31).
        int kB = kk + hi * 8;
#pragma unroll
        for (int t = 0; t < CT; ++t) {
            const unsigned short* bB = lWt + (size_t)(nCol0 + t * 16) * K + kB;
            BfVec bv;
            bv.q[0] = *(const uint4*)(bB);
            bv.q[1] = *(const uint4*)(bB + 16);
            acc[t] = __builtin_amdgcn_wmma_f32_16x16x32_bf16(
                         false, av.v, false, bv.v, (short)0, acc[t], false, false);
        }
    }

    // C/D layout: VGPR r -> row r (lanes 0-15) / row 8+r (lanes 16-31), col = lane&15
    int rowBase = rowTile * 16 + hi * 8;
#pragma unroll
    for (int t = 0; t < CT; ++t) {
        int nCol = nCol0 + t * 16;
        float bval = (nCol < Mreal) ? bias[nCol] : 0.0f;
#pragma unroll
        for (int r = 0; r < 8; ++r) {
            float v = acc[t][r] + bval;
            if (relu) v = v > 0.0f ? v : 0.0f;
            int row = rowBase + r;
            if (out32) {
                if (nCol < Mreal) out32[(size_t)row * Mreal + nCol] = v;
            } else {
                outb[(size_t)row * Mreal + nCol] = f2bf(v);
            }
        }
    }
}

// ---------------------------------------------------------------------------
// GAT per-layer scratch init: agg[n,j] = gat_bias[j]; denom = 0; mx = -inf(ordered)
// ---------------------------------------------------------------------------
__global__ void init_gat(float* __restrict__ agg, float* __restrict__ denom,
                         unsigned* __restrict__ mx, const float* __restrict__ bias) {
    int i = blockIdx.x * blockDim.x + threadIdx.x;
    if (i >= NN * 256) return;
    agg[i] = bias[i & 255];
    if (i < NN * 4) {
        denom[i] = 0.0f;
        mx[i] = f2ord(-3.0e38f);
    }
}

// ---------------------------------------------------------------------------
// Edge pass 1: logits[e,h] = sum_d att[h,d] * leaky(xl[src,h,d] + xr[dst,h,d]);
// segment-max via ordered-uint atomicMax.
// ---------------------------------------------------------------------------
__global__ void edge_logits(const unsigned short* __restrict__ xl,
                            const unsigned short* __restrict__ xr,
                            const float* __restrict__ att,
                            const int* __restrict__ src, const int* __restrict__ dst,
                            float* __restrict__ logits, unsigned* __restrict__ mx) {
    int idx = blockIdx.x * blockDim.x + threadIdx.x;      // e*4 + h
    if (idx >= NE * 4) return;
    int e = idx >> 2, h = idx & 3;
    int s = src[e], d = dst[e];
    const uint4* pl = (const uint4*)(xl + ((size_t)s * 256 + h * 64));
    const uint4* pr = (const uint4*)(xr + ((size_t)d * 256 + h * 64));
    const float* pa = att + h * 64;
    float acc = 0.0f;
#pragma unroll
    for (int c = 0; c < 8; ++c) {
        uint4 ul = pl[c], ur = pr[c];
        unsigned lu[4] = {ul.x, ul.y, ul.z, ul.w};
        unsigned ru[4] = {ur.x, ur.y, ur.z, ur.w};
#pragma unroll
        for (int i = 0; i < 4; ++i) {
            float a0 = bf2f(lu[i] & 0xFFFFu) + bf2f(ru[i] & 0xFFFFu);
            float a1 = bf2f(lu[i] >> 16)     + bf2f(ru[i] >> 16);
            a0 = a0 > 0.0f ? a0 : 0.2f * a0;
            a1 = a1 > 0.0f ? a1 : 0.2f * a1;
            int j = c * 8 + i * 2;
            acc += a0 * pa[j] + a1 * pa[j + 1];
        }
    }
    logits[idx] = acc;
    atomicMax(mx + ((size_t)d * 4 + h), f2ord(acc));
}

// Edge pass 2: ex = exp(logit - mx[dst,h]); denom[dst,h] += ex
__global__ void edge_exp(const float* __restrict__ logits, const unsigned* __restrict__ mx,
                         const int* __restrict__ dst,
                         float* __restrict__ exb, float* __restrict__ denom) {
    int idx = blockIdx.x * blockDim.x + threadIdx.x;
    if (idx >= NE * 4) return;
    int e = idx >> 2, h = idx & 3;
    int d = dst[e];
    float v = __expf(logits[idx] - ord2f(mx[(size_t)d * 4 + h]));
    exb[idx] = v;
    atomicAdd(denom + ((size_t)d * 4 + h), v);
}

// Edge pass 3: agg[dst,h,d] += (ex/denom[dst,h]) * xl[src,h,d]
// 64 threads per (edge, head); 4 groups per 256-thread block.
__global__ void edge_aggregate(const unsigned short* __restrict__ xl,
                               const float* __restrict__ exb,
                               const float* __restrict__ denom,
                               const int* __restrict__ src, const int* __restrict__ dst,
                               float* __restrict__ agg) {
    int gid = blockIdx.x * (blockDim.x >> 6) + (threadIdx.x >> 6);   // (e,h) index
    int d64 = threadIdx.x & 63;
    if (gid >= NE * 4) return;
    int e = gid >> 2, h = gid & 3;
    int s = src[e], dd = dst[e];
    float alpha = exb[gid] / denom[(size_t)dd * 4 + h];
    float v = alpha * bf2f((unsigned)xl[(size_t)s * 256 + h * 64 + d64]);
    atomicAdd(agg + ((size_t)dd * 256 + h * 64 + d64), v);
}

// ---------------------------------------------------------------------------
// LayerNorm over 64 features: one wave per node, 2 elems/lane, shfl_xor reduce.
// ---------------------------------------------------------------------------
__global__ void layernorm_bf16(const unsigned short* __restrict__ in,
                               const float* __restrict__ gamma,
                               const float* __restrict__ beta,
                               unsigned short* __restrict__ out) {
    int wave = threadIdx.x >> 5, lane = threadIdx.x & 31;
    int node = blockIdx.x * (blockDim.x >> 5) + wave;
    if (node >= NN) return;                              // wave-uniform
    const unsigned* p = (const unsigned*)(in + (size_t)node * 64);
    unsigned u = p[lane];
    float a = bf2f(u & 0xFFFFu), b = bf2f(u >> 16);
    float s = a + b;
#pragma unroll
    for (int m = 16; m >= 1; m >>= 1) s += __shfl_xor(s, m, 32);
    float mu = s * (1.0f / 64.0f);
    float da = a - mu, db = b - mu;
    float q = da * da + db * db;
#pragma unroll
    for (int m = 16; m >= 1; m >>= 1) q += __shfl_xor(q, m, 32);
    float rstd = rsqrtf(q * (1.0f / 64.0f) + 1e-5f);
    int j = lane * 2;
    float o0 = da * rstd * gamma[j]     + beta[j];
    float o1 = db * rstd * gamma[j + 1] + beta[j + 1];
    ((unsigned*)(out + (size_t)node * 64))[lane] =
        (unsigned)f2bf(o0) | ((unsigned)f2bf(o1) << 16);
}

// ---------------------------------------------------------------------------
// Host-side orchestration
// ---------------------------------------------------------------------------
extern "C" void kernel_launch(void* const* d_in, const int* in_sizes, int n_in,
                              void* d_out, int out_size, void* d_ws, size_t ws_size,
                              hipStream_t stream) {
    (void)in_sizes; (void)n_in; (void)out_size; (void)ws_size;
    typedef unsigned short u16;

    // --- input map (setup_inputs insertion order, depth-first) ---
    const float* x      = (const float*)d_in[0];                 // [NN,24,16]
    const float* W0     = (const float*)d_in[1];   const float* b0  = (const float*)d_in[2];
    const float* W1     = (const float*)d_in[3];   const float* b1  = (const float*)d_in[4];
    // gat layers at base 5 and 15: Wl,bl,Wr,br,att,bias,Wm,bm,gamma,beta
    const float* Wo0    = (const float*)d_in[25];  const float* bo0 = (const float*)d_in[26];
    const float* Wo1    = (const float*)d_in[27];  const float* bo1 = (const float*)d_in[28];
    const float* Wo2    = (const float*)d_in[29];  const float* bo2 = (const float*)d_in[30];
    const int*   src    = (const int*)d_in[31];                  // edge_index[0]
    const int*   dst    = src + NE;                              // edge_index[1]
    float* out = (float*)d_out;                                  // [NN,12]

    // --- workspace carve-out (256B aligned) ---
    size_t off = 0;
    auto alloc = [&](size_t bytes) -> char* {
        char* p = (char*)d_ws + off;
        off += (bytes + 255) & ~(size_t)255;
        return p;
    };
    u16*      xbf    = (u16*)alloc((size_t)NN * 384 * 2);
    u16*      hA     = (u16*)alloc((size_t)NN * 64 * 2);
    u16*      hB     = (u16*)alloc((size_t)NN * 64 * 2);
    u16*      xlb    = (u16*)alloc((size_t)NN * 256 * 2);
    u16*      xrb    = (u16*)alloc((size_t)NN * 256 * 2);
    float*    logits = (float*)alloc((size_t)NE * 4 * 4);
    float*    exb    = (float*)alloc((size_t)NE * 4 * 4);
    unsigned* mx     = (unsigned*)alloc((size_t)NN * 4 * 4);
    float*    denom  = (float*)alloc((size_t)NN * 4 * 4);
    float*    agg    = (float*)alloc((size_t)NN * 256 * 4);
    u16*      aggbf  = (u16*)alloc((size_t)NN * 256 * 2);
    u16*      Wt0    = (u16*)alloc((size_t)64  * 384 * 2);
    u16*      Wt1    = (u16*)alloc((size_t)64  * 64 * 2);
    u16*      WtL[2]; u16* WtR[2]; u16* WtM[2];
    for (int l = 0; l < 2; ++l) {
        WtL[l] = (u16*)alloc((size_t)256 * 64 * 2);
        WtR[l] = (u16*)alloc((size_t)256 * 64 * 2);
        WtM[l] = (u16*)alloc((size_t)64 * 256 * 2);
    }
    u16* WtO0 = (u16*)alloc((size_t)64 * 64 * 2);
    u16* WtO1 = (u16*)alloc((size_t)64 * 64 * 2);
    u16* WtO2 = (u16*)alloc((size_t)16 * 64 * 2);   // padded 12 -> 16 cols

    const int T = 256;
    auto gblk = [](long n, int t) { return (int)((n + t - 1) / t); };

    // --- weight transpose + convert (cheap, deterministic each call) ---
    wcvt_t<<<gblk(64 * 384, T), T, 0, stream>>>(W0, Wt0, 384, 64, 64);
    wcvt_t<<<gblk(64 * 64,  T), T, 0, stream>>>(W1, Wt1, 64, 64, 64);
    for (int l = 0; l < 2; ++l) {
        int base = 5 + 10 * l;
        wcvt_t<<<gblk(256 * 64, T), T, 0, stream>>>((const float*)d_in[base + 0], WtL[l], 64, 256, 256);
        wcvt_t<<<gblk(256 * 64, T), T, 0, stream>>>((const float*)d_in[base + 2], WtR[l], 64, 256, 256);
        wcvt_t<<<gblk(64 * 256, T), T, 0, stream>>>((const float*)d_in[base + 6], WtM[l], 256, 64, 64);
    }
    wcvt_t<<<gblk(64 * 64, T), T, 0, stream>>>(Wo0, WtO0, 64, 64, 64);
    wcvt_t<<<gblk(64 * 64, T), T, 0, stream>>>(Wo1, WtO1, 64, 64, 64);
    wcvt_t<<<gblk(16 * 64, T), T, 0, stream>>>(Wo2, WtO2, 64, 12, 16);

    // --- input convert: x [NN, 384] fp32 -> bf16 ---
    cvt_f32_bf16<<<gblk((long)NN * 384, T), T, 0, stream>>>(x, xbf, NN * 384);

    // GEMM launch helper: CT=4 col tiles per wave (CT=1 for the final layer),
    // dynamic LDS = whole transposed weight matrix.
    auto launch_g4 = [&](const u16* A, const u16* Wt, const float* bias, u16* ob,
                         float* o32, int K, int Mtiles, int Mreal, int relu) {
        int colGroups = Mtiles / 4;
        int grid = gblk((long)RT * colGroups, 8);          // 8 waves per block
        size_t sh = (size_t)Mtiles * 16 * K * 2;
        gemm_bf16_wmma<4><<<grid, T, sh, stream>>>(A, Wt, bias, ob, o32, K, Mtiles, Mreal, relu);
    };

    // --- input MLP: 384 -> 64 -> 64 (relu) ---
    launch_g4(xbf, Wt0, b0, hA, nullptr, 384, 4, 64, 1);
    launch_g4(hA,  Wt1, b1, hB, nullptr, 64,  4, 64, 1);
    u16* h = hB;

    // --- 2 GATv2 layers ---
    for (int l = 0; l < 2; ++l) {
        int base = 5 + 10 * l;
        const float* bl    = (const float*)d_in[base + 1];
        const float* br    = (const float*)d_in[base + 3];
        const float* att   = (const float*)d_in[base + 4];
        const float* gbias = (const float*)d_in[base + 5];
        const float* bm    = (const float*)d_in[base + 7];
        const float* gam   = (const float*)d_in[base + 8];
        const float* bet   = (const float*)d_in[base + 9];

        launch_g4(h, WtL[l], bl, xlb, nullptr, 64, 16, 256, 0);
        launch_g4(h, WtR[l], br, xrb, nullptr, 64, 16, 256, 0);

        init_gat<<<gblk((long)NN * 256, T), T, 0, stream>>>(agg, denom, mx, gbias);
        edge_logits<<<gblk((long)NE * 4, T), T, 0, stream>>>(xlb, xrb, att, src, dst, logits, mx);
        edge_exp<<<gblk((long)NE * 4, T), T, 0, stream>>>(logits, mx, dst, exb, denom);
        edge_aggregate<<<gblk((long)NE * 4 * 64, T), T, 0, stream>>>(xlb, exb, denom, src, dst, agg);

        cvt_f32_bf16<<<gblk((long)NN * 256, T), T, 0, stream>>>(agg, aggbf, NN * 256);
        launch_g4(aggbf, WtM[l], bm, hA, nullptr, 256, 4, 64, 1);
        layernorm_bf16<<<gblk((long)NN, 8), T, 0, stream>>>(hA, gam, bet, hB);
        h = hB;
    }

    // --- output MLP: 64 -> 64 -> 64 (relu) -> 12 ---
    launch_g4(h,  WtO0, bo0, hA, nullptr, 64, 4, 64, 1);
    launch_g4(hA, WtO1, bo1, hB, nullptr, 64, 4, 64, 1);
    {
        int grid = gblk((long)RT, 8);
        gemm_bf16_wmma<1><<<grid, T, (size_t)16 * 64 * 2, stream>>>(
            hB, WtO2, bo2, nullptr, out, 64, 1, 12, 0);
    }
}
